// muilt_view_10242_ori_5626407157943
// MI455X (gfx1250) — matrix operand order repaired and checked by reading, not compiled, not run on hardware
//
#include <hip/hip_runtime.h>
#include <hip/hip_bf16.h>
#include <math.h>
#include <stddef.h>

typedef __attribute__((ext_vector_type(16))) _Float16 v16h;
typedef __attribute__((ext_vector_type(8)))  _Float16 v8h;
typedef __attribute__((ext_vector_type(8)))  float    v8f;
typedef _Float16 h16;

#define BATCH 32

// ---------------- output offsets (flat fp32) ----------------
#define OFF_CLS      0           // (32,2)
#define OFF_NEW162   64          // (32,256,324)
#define OFF_MIN162   2654272
#define OFF_ATT162   5308480     // (32,1,324)
#define OFF_NEW10242 5318848     // (32,32,20484)
#define OFF_MIN10242 26294464
#define OFF_ATT10242 47270080    // (32,1,20484)

// ---------------- elementwise helpers ----------------
__global__ void k_zero(float* p, size_t n){
  size_t i = (size_t)blockIdx.x*blockDim.x + threadIdx.x;
  if (i < n) p[i] = 0.f;
}

__global__ void k_copy(const float* __restrict__ s, float* __restrict__ d, size_t n){
  size_t i = (size_t)blockIdx.x*blockDim.x + threadIdx.x;
  if (i < n) d[i] = s[i];
}

// Pack W [N x K] fp32 row-major -> f16 panel, K-block-major, with kl = k&31 permuted:
//   dst[((k>>5)*Np + n)*32 + ((kl>>3)&1)*16 + ((kl>>4)<<3) + (kl&7)] = W[n][k]   (zero padded)
// so that a B fragment (16 K in table order within a 32-block, fixed n) is
// two aligned 16B global_load_b128.
__global__ void k_pack_bt(const float* __restrict__ W, h16* __restrict__ dst,
                          int N, int K, int Np, int Kp){
  int i = blockIdx.x*blockDim.x + threadIdx.x;
  int tot = Kp*Np; if (i >= tot) return;
  int n = i % Np, k = i / Np;
  float v = (k < K && n < N) ? W[(size_t)n*K + k] : 0.f;
  dst[(((size_t)(k>>5))*Np + n)*32 + (((k>>3)&1)*16 + (((k&31)>>4)<<3) + (k&7))] = (h16)v;
}

// (B,C,N) channel-major -> point-major [B*N x C]
__global__ void k_chw_to_pm(const float* __restrict__ x, float* __restrict__ y, int C, int N){
  size_t i = (size_t)blockIdx.x*blockDim.x + threadIdx.x;
  size_t tot = (size_t)BATCH*N*C; if (i >= tot) return;
  int c = (int)(i % C); size_t t = i / C; int n = (int)(t % N); int b = (int)(t / N);
  y[i] = x[((size_t)b*C + c)*N + n];
}

// ---------------- WMMA GEMM: fused gather-7 variant ----------------
// Block = 128 threads (4 waves). Output tile: 16 rows x 64 cols
// (wave w handles cols [blockIdx.y*64 + w*16, +16)).
// A(row,k) = X[(b*Npts + neigh[n*7 + k/Cin])*Cin + k%Cin], row = b*Npts+n.
// A tile [16 x 32] f16 staged in LDS each K-step; B is the packed panel.
__global__ void k_wmma_gemm_gather(const float* __restrict__ X, const int* __restrict__ neigh,
                                   int Cin, int cshift, int Npts,
                                   const h16* __restrict__ Bw, const float* __restrict__ bias,
                                   float* __restrict__ Out,
                                   int M, int K, int Kpad, int Npad, int Nout){
  __shared__ h16      sa[16*32];     // A tile, row-major [16][32]
  __shared__ unsigned nbase[16*7];   // per-row gathered base offsets (elements)

  const int t     = threadIdx.x;        // 0..127
  const int lane  = t & 31;
  const int wave  = t >> 5;
  const int h     = lane >> 4;
  const int lo    = lane & 15;
  const int rowT  = blockIdx.x*16;

  // precompute neighbor base offsets for the 16 rows of this tile
  if (t < 112){
    int r = t / 7, j = t - r*7;
    int row = rowT + r;
    int b = row / Npts, n = row - b*Npts;
    nbase[t] = (unsigned)(((size_t)b*Npts + neigh[n*7 + j])*Cin);
  }
  __syncthreads();

  const int colT = blockIdx.y*64 + wave*16;
  const bool colActive = (colT < Npad);
  const int col = colT + lo;

  v8f acc = {0.f,0.f,0.f,0.f,0.f,0.f,0.f,0.f};

  for (int kk = 0; kk < Kpad; kk += 32){
    // ---- stage A tile: 512 elements, 4 per thread ----
    #pragma unroll
    for (int q = 0; q < 4; ++q){
      int e    = t + q*128;
      int r    = e >> 5;
      int kloc = e & 31;
      int k    = kk + kloc;
      float v = 0.f;
      if (k < K){
        int j, c;
        if (cshift >= 0){ j = k >> cshift; c = k & (Cin-1); }
        else            { j = k / Cin;     c = k - j*Cin;   }
        v = X[(size_t)nbase[r*7 + j] + c];
      }
      sa[r*32 + kloc] = (h16)v;
    }
    __syncthreads();

    if (colActive){
      // A fragment: two contiguous 8-half runs per lane
      v8h a0 = *(const v8h*)(&sa[lo*32 + h*8]);
      v8h a1 = *(const v8h*)(&sa[lo*32 + 16 + h*8]);
      v16h af, bf;
      #pragma unroll
      for (int i = 0; i < 8; ++i){ af[i] = a0[i]; af[i+8] = a1[i]; }
      // B fragment: 16 contiguous halfs in the packed panel
      const v8h* bp = (const v8h*)(Bw + (((size_t)(kk>>5))*Npad + col)*32 + h*16);
      v8h b0 = bp[0], b1 = bp[1];
      #pragma unroll
      for (int i = 0; i < 8; ++i){ bf[i] = b0[i]; bf[i+8] = b1[i]; }
      acc = __builtin_amdgcn_wmma_f32_16x16x32_f16(false, af, false, bf,
                                                   (short)0, acc, false, false);
    }
    __syncthreads();
  }

  if (colActive && col < Nout){
    float bb = bias ? bias[col] : 0.f;
    #pragma unroll
    for (int vv = 0; vv < 8; ++vv){
      int r = rowT + h*8 + vv;
      if (r < M) Out[(size_t)r*Nout + col] = acc[vv] + bb;
    }
  }
}

// ---------------- WMMA GEMM: dense variant (K % 32 == 0) ----------------
// One wave per block; 16x16 output tile; A rows contiguous fp32.
__global__ void k_wmma_gemm_direct(const float* __restrict__ X,
                                   const h16* __restrict__ Bw, const float* __restrict__ bias,
                                   float* __restrict__ Out,
                                   int M, int K, int Npad, int Nout){
  const int lane = threadIdx.x;
  const int h    = lane >> 4;
  const int lo   = lane & 15;
  const int row  = blockIdx.x*16 + lo;
  const int col  = blockIdx.y*16 + lo;
  const float* xrow = X + (size_t)row*K;

  v8f acc = {0.f,0.f,0.f,0.f,0.f,0.f,0.f,0.f};
  for (int kk = 0; kk < K; kk += 32){
    const float4* ap0 = (const float4*)(xrow + kk + h*8);        // K = kk+h*8 ..
    const float4* ap1 = (const float4*)(xrow + kk + 16 + h*8);   // K = kk+16+h*8 ..
    float4 a0 = ap0[0], a1 = ap0[1], a2 = ap1[0], a3 = ap1[1];
    v16h af, bf;
    af[0]=(h16)a0.x;  af[1]=(h16)a0.y;  af[2]=(h16)a0.z;  af[3]=(h16)a0.w;
    af[4]=(h16)a1.x;  af[5]=(h16)a1.y;  af[6]=(h16)a1.z;  af[7]=(h16)a1.w;
    af[8]=(h16)a2.x;  af[9]=(h16)a2.y;  af[10]=(h16)a2.z; af[11]=(h16)a2.w;
    af[12]=(h16)a3.x; af[13]=(h16)a3.y; af[14]=(h16)a3.z; af[15]=(h16)a3.w;
    const v8h* bp = (const v8h*)(Bw + (((size_t)(kk>>5))*Npad + col)*32 + h*16);
    v8h b0 = bp[0], b1 = bp[1];
    #pragma unroll
    for (int i = 0; i < 8; ++i){ bf[i] = b0[i]; bf[i+8] = b1[i]; }
    acc = __builtin_amdgcn_wmma_f32_16x16x32_f16(false, af, false, bf,
                                                 (short)0, acc, false, false);
  }
  if (col < Nout){
    float bb = bias ? bias[col] : 0.f;
    #pragma unroll
    for (int vv = 0; vv < 8; ++vv){
      int r = blockIdx.x*16 + h*8 + vv;
      if (r < M) Out[(size_t)r*Nout + col] = acc[vv] + bb;
    }
  }
}

// ---------------- BatchNorm (population stats over rows, per channel) ----------------
__global__ void k_bn_stats(const float* __restrict__ x, float* __restrict__ st,
                           size_t rows, int C){
  int c = blockIdx.x;
  float s = 0.f, s2 = 0.f;
  for (size_t r = (size_t)blockIdx.y*blockDim.x + threadIdx.x; r < rows;
       r += (size_t)blockDim.x*gridDim.y){
    float v = x[r*C + c]; s += v; s2 += v*v;
  }
  __shared__ float sh[256], sh2[256];
  sh[threadIdx.x] = s; sh2[threadIdx.x] = s2; __syncthreads();
  for (int o = 128; o > 0; o >>= 1){
    if ((int)threadIdx.x < o){ sh[threadIdx.x]+=sh[threadIdx.x+o]; sh2[threadIdx.x]+=sh2[threadIdx.x+o]; }
    __syncthreads();
  }
  if (threadIdx.x == 0){ atomicAdd(&st[2*c], sh[0]); atomicAdd(&st[2*c+1], sh2[0]); }
}

__global__ void k_bn_apply(const float* __restrict__ x, float* __restrict__ y,
                           const float* __restrict__ st,
                           const float* __restrict__ g, const float* __restrict__ be,
                           size_t rows, int C){
  size_t i = (size_t)blockIdx.x*blockDim.x + threadIdx.x;
  size_t tot = rows*C; if (i >= tot) return;
  int c = (int)(i % C);
  float inv = 1.f/(float)rows;
  float m = st[2*c]*inv;
  float var = st[2*c+1]*inv - m*m;
  float v = g[c]*(x[i]-m)*rsqrtf(var + 1e-5f) + be[c];
  y[i] = fmaxf(v, 0.f);
}

// ---------------- pool (mean of 7 gathered neighbors) ----------------
__global__ void k_pool(const float* __restrict__ x, const int* __restrict__ neigh,
                       float* __restrict__ out, int N, int Nc, int C){
  size_t i = (size_t)blockIdx.x*blockDim.x + threadIdx.x;
  size_t tot = (size_t)BATCH*Nc*C; if (i >= tot) return;
  int c = (int)(i % C); size_t t = i / C; int nc = (int)(t % Nc); int b = (int)(t / Nc);
  float s = 0.f;
  #pragma unroll
  for (int j = 0; j < 7; ++j)
    s += x[((size_t)(b*N + neigh[nc*7 + j]))*C + c];
  out[i] = s * (1.f/7.f);
}

// ---------------- half extract / insert along point dimension ----------------
__global__ void k_take_half(const float* __restrict__ src, float* __restrict__ dst,
                            int Nn, int C, int side){
  size_t i = (size_t)blockIdx.x*blockDim.x + threadIdx.x;
  size_t tot = (size_t)BATCH*Nn*C; if (i >= tot) return;
  int c = (int)(i % C); size_t r = i / C; int b = (int)(r / Nn); int n = (int)(r % Nn);
  dst[i] = src[((size_t)(b*2*Nn + side*Nn + n))*C + c];
}

__global__ void k_put_half(const float* __restrict__ src, float* __restrict__ dst,
                           int Nn, int C, int side){
  size_t i = (size_t)blockIdx.x*blockDim.x + threadIdx.x;
  size_t tot = (size_t)BATCH*Nn*C; if (i >= tot) return;
  int c = (int)(i % C); size_t r = i / C; int b = (int)(r / Nn); int n = (int)(r % Nn);
  dst[((size_t)(b*2*Nn + side*Nn + n))*C + c] = src[i];
}

// ---------------- attention (L=324, D=42, C=256) ----------------
__global__ void k_attn_scores(const float* __restrict__ q, const float* __restrict__ k,
                              float* __restrict__ sc, int L, int D){
  int row = blockIdx.x;           // b*L + l
  int b = row / L;
  const float* qp = q + (size_t)row*D;
  for (int m = threadIdx.x; m < L; m += blockDim.x){
    const float* kp = k + (size_t)(b*L + m)*D;
    float s = 0.f;
    for (int d = 0; d < D; ++d) s += qp[d]*kp[d];
    sc[(size_t)row*L + m] = s;
  }
}

__global__ void k_row_softmax(float* __restrict__ x, int W){
  __shared__ float sh[128];
  float* p = x + (size_t)blockIdx.x*W;
  int t = threadIdx.x;
  float mx = -1e30f;
  for (int i = t; i < W; i += 128) mx = fmaxf(mx, p[i]);
  sh[t] = mx; __syncthreads();
  for (int o = 64; o > 0; o >>= 1){ if (t < o) sh[t] = fmaxf(sh[t], sh[t+o]); __syncthreads(); }
  mx = sh[0]; __syncthreads();
  float s = 0.f;
  for (int i = t; i < W; i += 128){ float e = expf(p[i]-mx); p[i] = e; s += e; }
  sh[t] = s; __syncthreads();
  for (int o = 64; o > 0; o >>= 1){ if (t < o) sh[t] += sh[t+o]; __syncthreads(); }
  float inv = 1.f/sh[0]; __syncthreads();
  for (int i = t; i < W; i += 128) p[i] *= inv;
}

__global__ void k_attn_out(const float* __restrict__ sc, const float* __restrict__ v,
                           float* __restrict__ z, const float* __restrict__ gp, int L){
  int row = blockIdx.x; int b = row / L; int d = threadIdx.x;  // blockDim == 256 == C
  const float* sp = sc + (size_t)row*L;
  float s = 0.f;
  for (int m = 0; m < L; ++m) s += sp[m]*v[((size_t)(b*L + m))*256 + d];
  size_t zi = (size_t)row*256 + d;
  z[zi] = gp[0]*s + z[zi];
}

// ---------------- classifier pieces ----------------
__global__ void k_cls_att(const float* __restrict__ x, const float* __restrict__ aw,
                          const float* __restrict__ ab, float* __restrict__ a,
                          size_t rows, int C, int Nn, int twoNn, int aoff){
  size_t r = (size_t)blockIdx.x*blockDim.x + threadIdx.x;
  if (r >= rows) return;
  const float* xp = x + r*C;
  float s = ab[0];
  for (int c = 0; c < C; ++c) s += xp[c]*aw[c];
  int b = (int)(r / Nn), n = (int)(r % Nn);
  a[(size_t)b*twoNn + aoff + n] = 1.f/(1.f + expf(-s));
}

__global__ void k_scale_rows(const float* __restrict__ x, const float* __restrict__ a,
                             float* __restrict__ y, size_t rows, int C,
                             int Nn, int twoNn, int aoff){
  size_t i = (size_t)blockIdx.x*blockDim.x + threadIdx.x;
  size_t tot = rows*(size_t)C; if (i >= tot) return;
  size_t r = i / C; int b = (int)(r / Nn), n = (int)(r % Nn);
  y[i] = x[i]*a[(size_t)b*twoNn + aoff + n];
}

__global__ void k_mean_rows(const float* __restrict__ x, const float* __restrict__ a,
                            float* __restrict__ acc, int C, int Nn, int twoNn, int aoff){
  int c = blockIdx.x, b = blockIdx.y;
  float s = 0.f;
  for (int n = threadIdx.x; n < Nn; n += blockDim.x){
    float v = x[((size_t)(b*Nn + n))*C + c];
    if (a) v *= a[(size_t)b*twoNn + aoff + n];
    s += v;
  }
  __shared__ float sh[256];
  sh[threadIdx.x] = s; __syncthreads();
  for (int o = 128; o > 0; o >>= 1){ if ((int)threadIdx.x < o) sh[threadIdx.x] += sh[threadIdx.x+o]; __syncthreads(); }
  if (threadIdx.x == 0) acc[(size_t)b*C + c] += sh[0];
}

__global__ void k_cls_out(const float* __restrict__ acc, float cnt,
                          const float* __restrict__ fw, const float* __restrict__ fb,
                          float* __restrict__ out, int C, float wgt){
  int b = threadIdx.x; if (b >= BATCH) return;
  float inv = 1.f/cnt;
  float l0 = fb[0], l1 = fb[1];
  for (int c = 0; c < C; ++c){
    float m = acc[(size_t)b*C + c]*inv;
    l0 += m*fw[c]; l1 += m*fw[C + c];
  }
  float mx = fmaxf(l0, l1);
  float e0 = expf(l0-mx), e1 = expf(l1-mx);
  float s = 1.f/(e0+e1);
  out[b*2+0] += wgt*e0*s;
  out[b*2+1] += wgt*e1*s;
}

// ---------------- upconv top/down combine & concat ----------------
__global__ void k_upcombine(const float* __restrict__ y, const int* __restrict__ top,
                            const int* __restrict__ down, float* __restrict__ u,
                            int Nr, int Nn, int Cout){
  size_t i = (size_t)blockIdx.x*blockDim.x + threadIdx.x;
  size_t tot = (size_t)BATCH*Nn*Cout; if (i >= tot) return;
  int c = (int)(i % Cout); size_t t = i / Cout; int n = (int)(t % Nn); int b = (int)(t / Nn);
  const float* yb = y + (size_t)b*Nr*7*Cout;   // y_flat[b][p][o] contiguous
  float val;
  if (n < Nr) val = yb[(size_t)top[n]*Cout + c];
  else {
    int ii = n - Nr;
    val = 0.5f*(yb[(size_t)down[2*ii]*Cout + c] + yb[(size_t)down[2*ii+1]*Cout + c]);
  }
  u[i] = val;
}

__global__ void k_cat2(const float* __restrict__ u, const float* __restrict__ s,
                       float* __restrict__ out, size_t rows, int C1, int C2){
  size_t i = (size_t)blockIdx.x*blockDim.x + threadIdx.x;
  int Ct = C1 + C2;
  size_t tot = rows*(size_t)Ct; if (i >= tot) return;
  int c = (int)(i % Ct); size_t r = i / Ct;
  out[i] = (c < C1) ? u[r*C1 + c] : s[r*C2 + (c - C1)];
}

// ---------------- final output writers (channel-major) ----------------
__global__ void k_out162(const float* __restrict__ z, const float* __restrict__ xn,
                         float* __restrict__ out){
  size_t i = (size_t)blockIdx.x*blockDim.x + threadIdx.x;
  size_t tot = (size_t)32*256*324; if (i >= tot) return;
  int l = (int)(i % 324); size_t t = i / 324; int c = (int)(t % 256); int b = (int)(t / 256);
  size_t pm = ((size_t)(b*324 + l))*256 + c;
  float nv = xn[pm];
  out[OFF_NEW162 + i] = nv;
  out[OFF_MIN162 + i] = z[pm] - nv;
}

__global__ void k_out10242(const float* __restrict__ lb, const float* __restrict__ rb,
                           const float* __restrict__ a, float* __restrict__ out){
  size_t i = (size_t)blockIdx.x*blockDim.x + threadIdx.x;
  size_t tot = (size_t)32*32*20484; if (i >= tot) return;
  int l = (int)(i % 20484); size_t t = i / 20484; int c = (int)(t % 32); int b = (int)(t / 32);
  int side = l / 10242, n = l - side*10242;
  const float* src = side ? rb : lb;
  float x = src[((size_t)(b*10242 + n))*32 + c];
  float av = a[(size_t)b*20484 + l];
  out[OFF_NEW10242 + i] = x*av;
  out[OFF_MIN10242 + i] = x*(1.f - av);
}

// ============================================================
extern "C" void kernel_launch(void* const* d_in, const int* in_sizes, int n_in,
                              void* d_out, int out_size, void* d_ws, size_t ws_size,
                              hipStream_t stream){
  (void)in_sizes; (void)n_in; (void)out_size; (void)ws_size;
  float* out = (float*)d_out;

  // ---- inputs (setup_inputs dict order, recursively flattened) ----
  const float* X1 = (const float*)d_in[0];
  const float* X2 = (const float*)d_in[1];
  const float *d3w1=(const float*)d_in[2],*d3b1=(const float*)d_in[3],*d3g1=(const float*)d_in[4],*d3e1=(const float*)d_in[5];
  const float *d3w2=(const float*)d_in[6],*d3b2=(const float*)d_in[7],*d3g2=(const float*)d_in[8],*d3e2=(const float*)d_in[9];
  const float *d4w1=(const float*)d_in[10],*d4b1=(const float*)d_in[11],*d4g1=(const float*)d_in[12],*d4e1=(const float*)d_in[13];
  const float *d4w2=(const float*)d_in[14],*d4b2=(const float*)d_in[15],*d4g2=(const float*)d_in[16],*d4e2=(const float*)d_in[17];
  const float *d5w1=(const float*)d_in[18],*d5b1=(const float*)d_in[19],*d5g1=(const float*)d_in[20],*d5e1=(const float*)d_in[21];
  const float *d5w2=(const float*)d_in[22],*d5b2=(const float*)d_in[23],*d5g2=(const float*)d_in[24],*d5e2=(const float*)d_in[25];
  const float *d6w1=(const float*)d_in[26],*d6b1=(const float*)d_in[27],*d6g1=(const float*)d_in[28],*d6e1=(const float*)d_in[29];
  const float *d6w2=(const float*)d_in[30],*d6b2=(const float*)d_in[31],*d6g2=(const float*)d_in[32],*d6e2=(const float*)d_in[33];
  const float *wqp=(const float*)d_in[34],*bqp=(const float*)d_in[35];
  const float *wkp=(const float*)d_in[36],*bkp=(const float*)d_in[37];
  const float *wvp=(const float*)d_in[38],*bvp=(const float*)d_in[39];
  const float *gammap=(const float*)d_in[40];
  const float *aw162=(const float*)d_in[41],*ab162=(const float*)d_in[42],*fw162=(const float*)d_in[43],*fb162=(const float*)d_in[44];
  const float *aw642=(const float*)d_in[45],*ab642=(const float*)d_in[46],*fw642=(const float*)d_in[47],*fb642=(const float*)d_in[48];
  const float *aw2562=(const float*)d_in[49],*ab2562=(const float*)d_in[50],*fw2562=(const float*)d_in[51],*fb2562=(const float*)d_in[52];
  const float *aw10242=(const float*)d_in[53],*ab10242=(const float*)d_in[54],*fw10242=(const float*)d_in[55],*fb10242=(const float*)d_in[56];
  const float *uw5=(const float*)d_in[57],*ub5=(const float*)d_in[58];
  const float *u5w1=(const float*)d_in[59],*u5b1=(const float*)d_in[60],*u5g1=(const float*)d_in[61],*u5e1=(const float*)d_in[62];
  const float *u5w2=(const float*)d_in[63],*u5b2=(const float*)d_in[64],*u5g2=(const float*)d_in[65],*u5e2=(const float*)d_in[66];
  const float *uw4=(const float*)d_in[67],*ub4=(const float*)d_in[68];
  const float *u4w1=(const float*)d_in[69],*u4b1=(const float*)d_in[70],*u4g1=(const float*)d_in[71],*u4e1=(const float*)d_in[72];
  const float *u4w2=(const float*)d_in[73],*u4b2=(const float*)d_in[74],*u4g2=(const float*)d_in[75],*u4e2=(const float*)d_in[76];
  const float *uw3=(const float*)d_in[77],*ub3=(const float*)d_in[78];
  const float *u3w1=(const float*)d_in[79],*u3b1=(const float*)d_in[80],*u3g1=(const float*)d_in[81],*u3e1=(const float*)d_in[82];
  const float *u3w2=(const float*)d_in[83],*u3b2=(const float*)d_in[84],*u3g2=(const float*)d_in[85],*u3e2=(const float*)d_in[86];
  const float *endw=(const float*)d_in[87],*endb=(const float*)d_in[88];
  const int *n10242=(const int*)d_in[89],*n2562=(const int*)d_in[90],*n642=(const int*)d_in[91],*n162=(const int*)d_in[92];
  const int *t642=(const int*)d_in[93],*dn642=(const int*)d_in[94];
  const int *t2562=(const int*)d_in[95],*dn2562=(const int*)d_in[96];
  const int *t10242=(const int*)d_in[97],*dn10242=(const int*)d_in[98];

  // ---- workspace bump allocator (deterministic every call) ----
  char* wsb = (char*)d_ws; size_t wo = 0;
  auto alloc  = [&](size_t bytes)->void*{ wo=(wo+255)&~(size_t)255; void* p=wsb+wo; wo+=bytes; return p; };
  auto allocF = [&](size_t n)->float*{ return (float*)alloc(n*sizeof(float)); };

  float* xinL = allocF((size_t)327744*3);
  float* xinR = allocF((size_t)327744*3);
  float* x5l = allocF((size_t)327744*32); float* x5r = allocF((size_t)327744*32);
  float* x4l = allocF((size_t)81984*64);  float* x4r = allocF((size_t)81984*64);
  float* x3l = allocF((size_t)20544*128); float* x3r = allocF((size_t)20544*128);
  float* x2l = allocF((size_t)5184*256);  float* x2r = allocF((size_t)5184*256);
  float* tmpA   = allocF((size_t)327744*32);   // conv1 intermediates
  float* catbuf = allocF((size_t)327744*64);   // concat inputs / pooled / q,k,v alias
  float* ybuf   = allocF((size_t)81984*224);   // upconv GEMM out / scores alias
  float* ubuf   = allocF((size_t)327744*32);
  float* zb       = allocF((size_t)10368*256);
  float* new162pm = allocF((size_t)10368*256);
  float* l162in = allocF((size_t)5184*256);  float* r162in = allocF((size_t)5184*256);
  float* l642   = allocF((size_t)20544*128); float* r642   = allocF((size_t)20544*128);
  float* l2562  = allocF((size_t)81984*64);  float* r2562  = allocF((size_t)81984*64);
  float* l10242 = allocF((size_t)327744*32); float* r10242 = allocF((size_t)327744*32);
  float* a162 = allocF(10368); float* a642 = allocF(41088);
  float* a2562 = allocF(163968); float* a10242 = allocF(655488);
  float* stats = allocF(512); float* meanacc = allocF(8192);

  auto KPAD = [](int k){ return (k+31)&~31; };
  auto NPADF= [](int n){ return (n+15)&~15; };
  auto EG   = [](size_t n){ return dim3((unsigned)((n+255)/256)); };
  auto zero = [&](float* p, size_t n){ k_zero<<<EG(n),256,0,stream>>>(p,n); };

  // ---- pack all weights to f16 K-block-major panels ----
  auto packW = [&](const float* W, int N, int K)->const h16*{
    int Kp = KPAD(K), Np = NPADF(N);
    h16* dst = (h16*)alloc((size_t)Kp*Np*sizeof(h16));
    int tot = Kp*Np;
    k_pack_bt<<<dim3((tot+255)/256),256,0,stream>>>(W,dst,N,K,Np,Kp);
    return dst;
  };
  const h16 *h_d3w1=packW(d3w1,32,21),   *h_d3w2=packW(d3w2,32,224);
  const h16 *h_d4w1=packW(d4w1,64,224),  *h_d4w2=packW(d4w2,64,448);
  const h16 *h_d5w1=packW(d5w1,128,448), *h_d5w2=packW(d5w2,128,896);
  const h16 *h_d6w1=packW(d6w1,256,896), *h_d6w2=packW(d6w2,256,1792);
  const h16 *h_wq=packW(wqp,42,256), *h_wk=packW(wkp,42,256), *h_wv=packW(wvp,256,256);
  const h16 *h_uw5=packW(uw5,896,256), *h_u5w1=packW(u5w1,128,1792), *h_u5w2=packW(u5w2,128,896);
  const h16 *h_uw4=packW(uw4,448,128), *h_u4w1=packW(u4w1,64,896),   *h_u4w2=packW(u4w2,64,448);
  const h16 *h_uw3=packW(uw3,224,64),  *h_u3w1=packW(u3w1,32,448),   *h_u3w2=packW(u3w2,32,224);

  // ---- GEMM / BN / double-conv helpers ----
  auto gemmG = [&](const float* Xp, const int* ng, int Cin, int Npts,
                   const h16* Bw, const float* bias, float* Out, int Cout){
    int M = BATCH*Npts, K = 7*Cin, Kp = KPAD(K), Np = NPADF(Cout);
    int cshift = ((Cin & (Cin-1)) == 0) ? __builtin_ctz((unsigned)Cin) : -1;
    dim3 g((M+15)/16, (Np+63)/64);
    k_wmma_gemm_gather<<<g,128,0,stream>>>(Xp,ng,Cin,cshift,Npts,Bw,bias,Out,M,K,Kp,Np,Cout);
  };
  auto gemmD = [&](const float* Xp, int K, int M,
                   const h16* Bw, const float* bias, float* Out, int Nout){
    int Np = NPADF(Nout);                 // all direct calls: K % 32 == 0
    dim3 g((M+15)/16, Np/16);
    k_wmma_gemm_direct<<<g,32,0,stream>>>(Xp,Bw,bias,Out,M,K,Np,Nout);
  };
  auto bnrelu = [&](const float* x, float* y, size_t rows, int C,
                    const float* g, const float* be){
    zero(stats,(size_t)2*C);
    k_bn_stats<<<dim3(C,64),256,0,stream>>>(x,stats,rows,C);
    k_bn_apply<<<EG(rows*(size_t)C),256,0,stream>>>(x,y,stats,g,be,rows,C);
  };
  auto dconv = [&](const float* xin, const int* ng, int Npts, int Cin, int Cout,
                   const h16* w1h, const float* b1, const float* g1, const float* e1,
                   const h16* w2h, const float* b2, const float* g2, const float* e2,
                   float* outp){
    size_t M = (size_t)BATCH*Npts;
    gemmG(xin,ng,Cin,Npts,w1h,b1,tmpA,Cout);
    bnrelu(tmpA,tmpA,M,Cout,g1,e1);
    gemmG(tmpA,ng,Cout,Npts,w2h,b2,outp,Cout);
    bnrelu(outp,outp,M,Cout,g2,e2);
  };

  zero(out + OFF_CLS, 64);   // x_class accumulator

  // ================= encoder (both hemispheres) =================
  k_chw_to_pm<<<EG((size_t)327744*3),256,0,stream>>>(X1,xinL,3,10242);
  k_chw_to_pm<<<EG((size_t)327744*3),256,0,stream>>>(X2,xinR,3,10242);
  float* pooled = catbuf;
  auto encode = [&](const float* xin, float* x5, float* x4, float* x3, float* x2b){
    dconv(xin,n10242,10242,3,32,  h_d3w1,d3b1,d3g1,d3e1, h_d3w2,d3b2,d3g2,d3e2, x5);
    k_pool<<<EG((size_t)BATCH*2562*32),256,0,stream>>>(x5,n10242,pooled,10242,2562,32);
    dconv(pooled,n2562,2562,32,64, h_d4w1,d4b1,d4g1,d4e1, h_d4w2,d4b2,d4g2,d4e2, x4);
    k_pool<<<EG((size_t)BATCH*642*64),256,0,stream>>>(x4,n2562,pooled,2562,642,64);
    dconv(pooled,n642,642,64,128,  h_d5w1,d5b1,d5g1,d5e1, h_d5w2,d5b2,d5g2,d5e2, x3);
    k_pool<<<EG((size_t)BATCH*162*128),256,0,stream>>>(x3,n642,pooled,642,162,128);
    dconv(pooled,n162,162,128,256, h_d6w1,d6b1,d6g1,d6e1, h_d6w2,d6b2,d6g2,d6e2, x2b);
  };
  encode(xinL,x5l,x4l,x3l,x2l);
  encode(xinR,x5r,x4r,x3r,x2r);

  // ================= bottleneck: concat + self-attention =================
  k_put_half<<<EG((size_t)5184*256),256,0,stream>>>(x2l,zb,162,256,0);
  k_put_half<<<EG((size_t)5184*256),256,0,stream>>>(x2r,zb,162,256,1);
  float* qb = catbuf;
  float* kb = qb + (size_t)10368*42;
  float* vb = kb + (size_t)10368*42;
  gemmD(zb,256,10368,h_wq,bqp,qb,42);
  gemmD(zb,256,10368,h_wk,bkp,kb,42);
  gemmD(zb,256,10368,h_wv,bvp,vb,256);
  float* sc = ybuf;
  k_attn_scores<<<10368,128,0,stream>>>(qb,kb,sc,324,42);
  k_row_softmax<<<10368,128,0,stream>>>(sc,324);
  k_attn_out<<<10368,256,0,stream>>>(sc,vb,zb,gammap,324);

  // ---- classify 162 + its outputs ----
  k_cls_att<<<EG(10368),256,0,stream>>>(zb,aw162,ab162,a162,10368,256,324,324,0);
  k_scale_rows<<<EG((size_t)10368*256),256,0,stream>>>(zb,a162,new162pm,10368,256,324,324,0);
  zero(meanacc,(size_t)32*256);
  k_mean_rows<<<dim3(256,32),256,0,stream>>>(new162pm,nullptr,meanacc,256,324,324,0);
  k_cls_out<<<1,32,0,stream>>>(meanacc,324.f,fw162,fb162,out,256,0.2f);
  k_out162<<<EG((size_t)32*256*324),256,0,stream>>>(zb,new162pm,out);
  k_copy<<<EG(10368),256,0,stream>>>(a162,out+OFF_ATT162,10368);

  // ================= decoder =================
  auto upstage = [&](const float* xin_side, int Nr, int Cx, int Cout,
                     const h16* uwh, const float* ubias,
                     const int* ng, const int* top, const int* down, int Nn,
                     const float* skip,
                     const h16* w1h, const float* b1, const float* g1, const float* e1,
                     const h16* w2h, const float* b2, const float* g2, const float* e2,
                     float* out_side){
    gemmD(xin_side,Cx,BATCH*Nr,uwh,ubias,ybuf,7*Cout);
    k_upcombine<<<EG((size_t)BATCH*Nn*Cout),256,0,stream>>>(ybuf,top,down,ubuf,Nr,Nn,Cout);
    k_cat2<<<EG((size_t)BATCH*Nn*2*Cout),256,0,stream>>>(ubuf,skip,catbuf,(size_t)BATCH*Nn,Cout,Cout);
    dconv(catbuf,ng,Nn,2*Cout,Cout,w1h,b1,g1,e1,w2h,b2,g2,e2,out_side);
  };

  // -- up5: 162 -> 642 --
  k_take_half<<<EG((size_t)5184*256),256,0,stream>>>(new162pm,l162in,162,256,0);
  k_take_half<<<EG((size_t)5184*256),256,0,stream>>>(new162pm,r162in,162,256,1);
  upstage(l162in,162,256,128,h_uw5,ub5,n642,t642,dn642,642,x3l,
          h_u5w1,u5b1,u5g1,u5e1,h_u5w2,u5b2,u5g2,u5e2,l642);
  upstage(r162in,162,256,128,h_uw5,ub5,n642,t642,dn642,642,x3r,
          h_u5w1,u5b1,u5g1,u5e1,h_u5w2,u5b2,u5g2,u5e2,r642);
  // classify 642 (scale in place -> new642)
  k_cls_att<<<EG((size_t)20544),256,0,stream>>>(l642,aw642,ab642,a642,20544,128,642,1284,0);
  k_cls_att<<<EG((size_t)20544),256,0,stream>>>(r642,aw642,ab642,a642,20544,128,642,1284,642);
  k_scale_rows<<<EG((size_t)20544*128),256,0,stream>>>(l642,a642,l642,20544,128,642,1284,0);
  k_scale_rows<<<EG((size_t)20544*128),256,0,stream>>>(r642,a642,r642,20544,128,642,1284,642);
  zero(meanacc,(size_t)32*128);
  k_mean_rows<<<dim3(128,32),256,0,stream>>>(l642,nullptr,meanacc,128,642,1284,0);
  k_mean_rows<<<dim3(128,32),256,0,stream>>>(r642,nullptr,meanacc,128,642,1284,0);
  k_cls_out<<<1,32,0,stream>>>(meanacc,1284.f,fw642,fb642,out,128,0.2f);

  // -- up4: 642 -> 2562 --
  upstage(l642,642,128,64,h_uw4,ub4,n2562,t2562,dn2562,2562,x4l,
          h_u4w1,u4b1,u4g1,u4e1,h_u4w2,u4b2,u4g2,u4e2,l2562);
  upstage(r642,642,128,64,h_uw4,ub4,n2562,t2562,dn2562,2562,x4r,
          h_u4w1,u4b1,u4g1,u4e1,h_u4w2,u4b2,u4g2,u4e2,r2562);
  k_cls_att<<<EG((size_t)81984),256,0,stream>>>(l2562,aw2562,ab2562,a2562,81984,64,2562,5124,0);
  k_cls_att<<<EG((size_t)81984),256,0,stream>>>(r2562,aw2562,ab2562,a2562,81984,64,2562,5124,2562);
  k_scale_rows<<<EG((size_t)81984*64),256,0,stream>>>(l2562,a2562,l2562,81984,64,2562,5124,0);
  k_scale_rows<<<EG((size_t)81984*64),256,0,stream>>>(r2562,a2562,r2562,81984,64,2562,5124,2562);
  zero(meanacc,(size_t)32*64);
  k_mean_rows<<<dim3(64,32),256,0,stream>>>(l2562,nullptr,meanacc,64,2562,5124,0);
  k_mean_rows<<<dim3(64,32),256,0,stream>>>(r2562,nullptr,meanacc,64,2562,5124,0);
  k_cls_out<<<1,32,0,stream>>>(meanacc,5124.f,fw2562,fb2562,out,64,0.2f);

  // -- up3: 2562 -> 10242 --
  upstage(l2562,2562,64,32,h_uw3,ub3,n10242,t10242,dn10242,10242,x5l,
          h_u3w1,u3b1,u3g1,u3e1,h_u3w2,u3b2,u3g2,u3e2,l10242);
  upstage(r2562,2562,64,32,h_uw3,ub3,n10242,t10242,dn10242,10242,x5r,
          h_u3w1,u3b1,u3g1,u3e1,h_u3w2,u3b2,u3g2,u3e2,r10242);
  // classify 10242: keep x unscaled, fold gate into mean + output writer
  k_cls_att<<<EG((size_t)327744),256,0,stream>>>(l10242,aw10242,ab10242,a10242,327744,32,10242,20484,0);
  k_cls_att<<<EG((size_t)327744),256,0,stream>>>(r10242,aw10242,ab10242,a10242,327744,32,10242,20484,10242);
  zero(meanacc,(size_t)32*32);
  k_mean_rows<<<dim3(32,32),256,0,stream>>>(l10242,a10242,meanacc,32,10242,20484,0);
  k_mean_rows<<<dim3(32,32),256,0,stream>>>(r10242,a10242,meanacc,32,10242,20484,10242);
  k_cls_out<<<1,32,0,stream>>>(meanacc,20484.f,fw10242,fb10242,out,32,0.2f);
  k_cls_out<<<1,32,0,stream>>>(meanacc,20484.f,endw,endb,out,32,0.2f);   // x_end head

  k_out10242<<<EG((size_t)32*32*20484),256,0,stream>>>(l10242,r10242,a10242,out);
  k_copy<<<EG((size_t)655488),256,0,stream>>>(a10242,out+OFF_ATT10242,655488);
}
